// HGNNs_28424093565236
// MI455X (gfx1250) — matrix operand order repaired and checked by reading, not compiled, not run on hardware
//
#include <hip/hip_runtime.h>
#include <hip/hip_bf16.h>

#define NN 100000
#define EE 800000
#define TT 4
#define FF 128
#define HH 4
#define DD 32

typedef __attribute__((ext_vector_type(16))) __bf16 v16bf;
typedef __attribute__((ext_vector_type(8)))  float  v8f;

__device__ __forceinline__ __bf16 f2bf(float f) { return (__bf16)f; }

// acc = 4*x + sum_t bias[l][t]   (residual + bias folded across the 4 edge types)
__global__ __launch_bounds__(256) void init_acc_k(float* __restrict__ acc,
                                                  const float* __restrict__ x,
                                                  const float* __restrict__ bias_l) {
    int i = blockIdx.x * 256 + threadIdx.x;            // over N*F = 12.8M
    int f = i & (FF - 1);
    float b = bias_l[f] + bias_l[FF + f] + bias_l[2 * FF + f] + bias_l[3 * FF + f];
    acc[i] = 4.0f * x[i] + b;
}

// feat = x @ W  (N x 128 @ 128 x 128), bf16 WMMA, f32 accumulate.
// Block: 16 rows x 128 cols. 8 waves, one 16x16 WMMA tile per wave, 4 k-steps.
__global__ __launch_bounds__(256) void gemm_feat_k(const float* __restrict__ X,
                                                   const float* __restrict__ W,
                                                   float* __restrict__ feat) {
    const int m0   = blockIdx.x * 16;                  // N = 100000 = 6250*16, no tail
    const int wave = threadIdx.x >> 5;
    const int lane = threadIdx.x & 31;
    const int r    = lane & 15;
    const int half = lane >> 4;
    const int n0   = wave * 16;

    v8f c = {};
#pragma unroll
    for (int k0 = 0; k0 < FF; k0 += 32) {
        v16bf a, b;
        // A: 16x32 bf16. lane-half 0 -> K = k0+{0..15}; half 1 -> K = k0+{16..31}
        const float* xrow = X + (size_t)(m0 + r) * FF + k0 + half * 16;
#pragma unroll
        for (int j = 0; j < 8; ++j) a[j]     = f2bf(xrow[j]);
#pragma unroll
        for (int j = 0; j < 8; ++j) a[8 + j] = f2bf(xrow[8 + j]);
        // B: 32x16 bf16, row-striped across lanes: element j -> K = k0 + half*16 + j, N = n0 + r
        const float* wcol = W + (size_t)(k0 + half * 16) * FF + n0 + r;
#pragma unroll
        for (int j = 0; j < 16; ++j) b[j] = f2bf(wcol[(size_t)j * FF]);

        c = __builtin_amdgcn_wmma_f32_16x16x32_bf16(false, a, false, b,
                                                    (short)0, c, false, false);
    }
    // C/D: VGPR i -> row m0 + half*8 + i, col n0 + r
    float* out = feat + (size_t)m0 * FF + n0 + r;
#pragma unroll
    for (int i = 0; i < 8; ++i) out[(size_t)(half * 8 + i) * FF] = c[i];
}

// el[n][h] = sum_d feat[n][h*32+d]*al[h][d]; er likewise. One wave == one head (D==32).
// Also seeds m = -inf, denom = 0 for the softmax.
__global__ __launch_bounds__(256) void attn_logits_k(const float* __restrict__ feat,
                                                     const float* __restrict__ al,
                                                     const float* __restrict__ ar,
                                                     float* __restrict__ el,
                                                     float* __restrict__ er,
                                                     float* __restrict__ mbuf,
                                                     float* __restrict__ dnom) {
    int n = blockIdx.x * 2 + (threadIdx.x >> 7);       // 2 nodes / 256-thread block
    int f = threadIdx.x & (FF - 1);
    int head = f >> 5, d = f & 31;
    float v  = feat[(size_t)n * FF + f];
    float pl = v * al[head * DD + d];
    float pr = v * ar[head * DD + d];
#pragma unroll
    for (int off = 16; off; off >>= 1) {
        pl += __shfl_xor(pl, off, 32);
        pr += __shfl_xor(pr, off, 32);
    }
    if ((threadIdx.x & 31) == 0) {
        int idx = n * HH + head;
        el[idx] = pl;  er[idx] = pr;
        mbuf[idx] = -__builtin_inff();
        dnom[idx] = 0.0f;
    }
}

__device__ __forceinline__ void atomicMaxF(float* addr, float v) {
    if (v >= 0.0f) atomicMax((int*)addr, __float_as_int(v));
    else           atomicMin((unsigned int*)addr, __float_as_uint(v));
}

// Pass 1: e = leaky_relu(el[src]+er[dst]); segment max over dst via float atomic-max.
__global__ __launch_bounds__(256) void edge_logits_k(const int* __restrict__ src,
                                                     const int* __restrict__ dst,
                                                     const float* __restrict__ el,
                                                     const float* __restrict__ er,
                                                     float* __restrict__ ev,
                                                     float* __restrict__ mbuf) {
    int i = blockIdx.x * 256 + threadIdx.x;            // over E*H = 3.2M
    if (i >= EE * HH) return;
    int e = i >> 2, h = i & 3;
    int s = src[e], d = dst[e];
    float x = el[s * HH + h] + er[d * HH + h];
    x = (x > 0.0f) ? x : 0.2f * x;
    ev[i] = x;
    atomicMaxF(mbuf + d * HH + h, x);
}

// Pass 2: ex = exp(e - m[dst]); denom[dst] += ex.
__global__ __launch_bounds__(256) void edge_exp_k(const int* __restrict__ dst,
                                                  float* __restrict__ ev,
                                                  const float* __restrict__ mbuf,
                                                  float* __restrict__ dnom) {
    int i = blockIdx.x * 256 + threadIdx.x;
    if (i >= EE * HH) return;
    int e = i >> 2, h = i & 3;
    int d = dst[e];
    float ex = __expf(ev[i] - mbuf[d * HH + h]);
    ev[i] = ex;
    atomicAdd(dnom + d * HH + h, ex);
}

// Pass 3: acc[dst] += (ex/denom[dst]) * feat[src]. One wave per edge, lane == d in head.
__global__ __launch_bounds__(256) void edge_msg_k(const int* __restrict__ src,
                                                  const int* __restrict__ dst,
                                                  const float* __restrict__ ev,
                                                  const float* __restrict__ dnom,
                                                  const float* __restrict__ feat,
                                                  float* __restrict__ acc) {
    int e = blockIdx.x * 8 + (threadIdx.x >> 5);       // 8 waves/block, E = 800000
    if (e >= EE) return;
    int lane = threadIdx.x & 31;
    int s = src[e], d = dst[e];
#pragma unroll
    for (int h = 0; h < HH; ++h) {
        float alpha = ev[e * HH + h] / dnom[d * HH + h];
        float fv = feat[(size_t)s * FF + h * DD + lane];
        atomicAdd(acc + (size_t)d * FF + h * DD + lane, alpha * fv);
    }
}

__global__ __launch_bounds__(256) void elu_k(float* __restrict__ acc) {
    int i = blockIdx.x * 256 + threadIdx.x;            // over N*F
    float v = acc[i];
    acc[i] = (v > 0.0f) ? v : (__expf(v) - 1.0f);
}

extern "C" void kernel_launch(void* const* d_in, const int* in_sizes, int n_in,
                              void* d_out, int out_size, void* d_ws, size_t ws_size,
                              hipStream_t stream) {
    const float* poi   = (const float*)d_in[0];
    const int*   edges = (const int*)  d_in[1];   // [T][2][E]
    const float* Wall  = (const float*)d_in[2];   // [L][T][F][F]
    const float* alall = (const float*)d_in[3];   // [L][T][H][D]
    const float* arall = (const float*)d_in[4];
    const float* ball  = (const float*)d_in[5];   // [L][T][F]
    float* out = (float*)d_out;

    float* ws   = (float*)d_ws;
    float* feat = ws;                                   // N*F
    float* hbuf = feat + (size_t)NN * FF;               // N*F
    float* el   = hbuf + (size_t)NN * FF;               // N*H
    float* er   = el + (size_t)NN * HH;
    float* mb   = er + (size_t)NN * HH;
    float* dn   = mb + (size_t)NN * HH;
    float* ev   = dn + (size_t)NN * HH;                 // E*H

    const int gNF   = (NN * FF) / 256;                  // 50000
    const int gGemm = NN / 16;                          // 6250
    const int gAttn = NN / 2;                           // 50000
    const int gEH   = (EE * HH + 255) / 256;            // 12500
    const int gMsg  = (EE + 7) / 8;                     // 100000

    for (int l = 0; l < 2; ++l) {
        const float* x   = l ? hbuf : poi;
        float*       acc = l ? out  : hbuf;
        init_acc_k<<<gNF, 256, 0, stream>>>(acc, x, ball + (size_t)l * TT * FF);
        for (int t = 0; t < TT; ++t) {
            const float* Wt = Wall  + (size_t)(l * TT + t) * FF * FF;
            const float* al = alall + (size_t)(l * TT + t) * HH * DD;
            const float* ar = arall + (size_t)(l * TT + t) * HH * DD;
            const int* srcp = edges + (size_t)t * 2 * EE;
            const int* dstp = srcp + EE;
            gemm_feat_k  <<<gGemm, 256, 0, stream>>>(x, Wt, feat);
            attn_logits_k<<<gAttn, 256, 0, stream>>>(feat, al, ar, el, er, mb, dn);
            edge_logits_k<<<gEH,   256, 0, stream>>>(srcp, dstp, el, er, ev, mb);
            edge_exp_k   <<<gEH,   256, 0, stream>>>(dstp, ev, mb, dn);
            edge_msg_k   <<<gMsg,  256, 0, stream>>>(srcp, dstp, ev, dn, feat, acc);
        }
        if (l == 0) elu_k<<<gNF, 256, 0, stream>>>(acc);
    }
}
